// RecurrentDecoder_22505628631623
// MI455X (gfx1250) — matrix operand order repaired and checked by reading, not compiled, not run on hardware
//
#include <hip/hip_runtime.h>
#include <hip/hip_bf16.h>
#include <math.h>

// Problem constants (match reference)
#define BB 64
#define TT 32
#define SS 64
#define EMB 512
#define HH 1024
#define ENC 1024
#define VV 32000

typedef __attribute__((ext_vector_type(16))) __bf16 v16bf;
typedef __attribute__((ext_vector_type(8)))  float  v8f;

union FragU { uint4 u[2]; v16bf v; };

// ---------------------------------------------------------------------------
// Generic bf16 WMMA GEMM: C[M][N] (f32, row-major) = A[M][K] (bf16) x BT[N][K]^T
// Wave tile 64M x 64N, block = 4 waves covering 256 columns.
// Requires M%64==0, N%64==0 (grid covers exactly), K%32==0.
// A-fragment (16x32 MxK, bf16): lane L -> row L&15; K chunks at
//   k0 + (L>>4)*8 + [0..7]  (VGPR0-3) and k0 + 16 + (L>>4)*8 + [0..7] (VGPR4-7)
// B-fragment (32x16 KxN, bf16) from BT[N][K]: lane L -> col L&15; K at
//   k0 + (L>>4)*8 + [0..7] and k0 + 16 + (L>>4)*8 + [0..7] (same as A)
// ---------------------------------------------------------------------------
__global__ void __launch_bounds__(128)
gemm_bf16_tn(const __bf16* __restrict__ A, const __bf16* __restrict__ BT,
             float* __restrict__ C, int M, int N, int K)
{
    const int lane = threadIdx.x & 31;
    const int wave = threadIdx.x >> 5;
    const int half = lane >> 4;     // 0 for lanes 0-15, 1 for 16-31
    const int lr   = lane & 15;
    const int m0   = blockIdx.y * 64;
    const int n0   = (blockIdx.x * 4 + wave) * 64;
    if (n0 >= N || m0 >= M) return;

    v8f acc[4][4] = {};

    for (int k0 = 0; k0 < K; k0 += 32) {
        FragU fa[4];
#pragma unroll
        for (int ms = 0; ms < 4; ++ms) {
            const __bf16* ap = A + (size_t)(m0 + ms * 16 + lr) * K + k0 + half * 8;
            fa[ms].u[0] = *(const uint4*)(ap);        // K = k0+half*8   .. +7
            fa[ms].u[1] = *(const uint4*)(ap + 16);   // K = k0+16+half*8.. +7
        }
#pragma unroll
        for (int ns = 0; ns < 4; ++ns) {
            FragU fb;
            const __bf16* bp = BT + (size_t)(n0 + ns * 16 + lr) * K + k0 + half * 8;
            fb.u[0] = *(const uint4*)(bp);
            fb.u[1] = *(const uint4*)(bp + 16);
#pragma unroll
            for (int ms = 0; ms < 4; ++ms) {
                acc[ms][ns] = __builtin_amdgcn_wmma_f32_16x16x32_bf16(
                    false, fa[ms].v, false, fb.v, (short)0, acc[ms][ns],
                    false, false);
            }
        }
    }

    // C/D layout: VGPR r -> row r + half*8, col = lane&15 (per 16x16 subtile)
#pragma unroll
    for (int ms = 0; ms < 4; ++ms)
#pragma unroll
        for (int ns = 0; ns < 4; ++ns)
#pragma unroll
            for (int r = 0; r < 8; ++r) {
                int row = m0 + ms * 16 + r + half * 8;
                int col = n0 + ns * 16 + lr;
                C[(size_t)row * N + col] = acc[ms][ns][r];
            }
}

// ---------------------------------------------------------------------------
// Prep kernels
// ---------------------------------------------------------------------------
__global__ void __launch_bounds__(256)
convert_f32_bf16(const float* __restrict__ src, __bf16* __restrict__ dst, size_t n)
{
    size_t i = (size_t)blockIdx.x * blockDim.x + threadIdx.x;
    size_t stride = (size_t)gridDim.x * blockDim.x;
    for (; i < n; i += stride) dst[i] = (__bf16)src[i];
}

// src f32 [R][C] -> dst bf16 [C][R]; R,C multiples of 32
__global__ void __launch_bounds__(256)
transpose_f32_bf16(const float* __restrict__ src, __bf16* __restrict__ dst,
                   int R, int C)
{
    __shared__ float tile[32][33];
    int tx = threadIdx.x & 31;
    int ty = threadIdx.x >> 5;  // 0..7
    int x  = blockIdx.x * 32 + tx;
    int y0 = blockIdx.y * 32;
    for (int j = ty; j < 32; j += 8)
        tile[j][tx] = src[(size_t)(y0 + j) * C + x];
    __syncthreads();
    int x2 = blockIdx.y * 32 + tx;  // row index in src -> col in dst
    for (int j = ty; j < 32; j += 8)
        dst[(size_t)(blockIdx.x * 32 + j) * R + x2] = (__bf16)tile[tx][j];
}

// ---------------------------------------------------------------------------
// Per-step kernels
// ---------------------------------------------------------------------------
// x_bf16[B][EMB+H] = [emb_t, a]; h_bf16[B][H] = (bf16)h
__global__ void __launch_bounds__(256)
pack_x(const float* __restrict__ emb, const float* __restrict__ a,
       const float* __restrict__ h, __bf16* __restrict__ x_bf16,
       __bf16* __restrict__ h_bf16, int t)
{
    const int NX = BB * (EMB + HH);
    const int NT = NX + BB * HH;
    int idx = blockIdx.x * blockDim.x + threadIdx.x;
    if (idx >= NT) return;
    if (idx < NX) {
        int b = idx / (EMB + HH), k = idx % (EMB + HH);
        float v = (k < EMB) ? emb[((size_t)b * TT + t) * EMB + k]
                            : a[(size_t)b * HH + (k - EMB)];
        x_bf16[idx] = (__bf16)v;
    } else {
        int i2 = idx - NX;
        h_bf16[i2] = (__bf16)h[i2];
    }
}

// GRU gate fusion: h <- (1-z)*n + z*h ; also emit bf16 copy of h_new
__global__ void __launch_bounds__(256)
gru_update(const float* __restrict__ gi, const float* __restrict__ gh,
           const float* __restrict__ b_ih, const float* __restrict__ b_hh,
           float* __restrict__ h, __bf16* __restrict__ hn_bf16)
{
    int idx = blockIdx.x * blockDim.x + threadIdx.x;
    if (idx >= BB * HH) return;
    int b = idx / HH, j = idx % HH;
    const float* gib = gi + (size_t)b * 3 * HH;
    const float* ghb = gh + (size_t)b * 3 * HH;
    float ir = gib[j]          + b_ih[j];
    float iz = gib[HH + j]     + b_ih[HH + j];
    float in = gib[2 * HH + j] + b_ih[2 * HH + j];
    float hr = ghb[j]          + b_hh[j];
    float hz = ghb[HH + j]     + b_hh[HH + j];
    float hn = ghb[2 * HH + j] + b_hh[2 * HH + j];
    float r = 1.0f / (1.0f + expf(-(ir + hr)));
    float z = 1.0f / (1.0f + expf(-(iz + hz)));
    float n = tanhf(in + r * hn);
    float hv = (1.0f - z) * n + z * h[idx];
    h[idx] = hv;
    hn_bf16[idx] = (__bf16)hv;
}

// scores[b][s] = sum_h tanh(projk[b,s,h] + pq[b,h]) * ev[h], masked.
// one wave per (b,s) pair; 8 waves per block.
__global__ void __launch_bounds__(256)
energy_scores(const float* __restrict__ projk, const float* __restrict__ pq,
              const float* __restrict__ ev, const unsigned char* __restrict__ mask,
              float* __restrict__ scores)
{
    int lane = threadIdx.x & 31;
    int widx = (blockIdx.x * blockDim.x + threadIdx.x) >> 5;  // 0..B*S-1
    int b = widx >> 6;  // /S
    int s = widx & 63;  // %S
    const float* pk = projk + ((size_t)b * SS + s) * HH;
    const float* q  = pq + (size_t)b * HH;
    float sum = 0.0f;
    for (int hh = lane; hh < HH; hh += 32)
        sum += tanhf(pk[hh] + q[hh]) * ev[hh];
#pragma unroll
    for (int off = 16; off > 0; off >>= 1)
        sum += __shfl_xor(sum, off, 32);
    if (lane == 0)
        scores[widx] = mask[widx] ? sum : -1e9f;
}

// per-b: softmax over S, write att_probs, compute context[b][e]
__global__ void __launch_bounds__(256)
softmax_context(const float* __restrict__ scores, const float* __restrict__ enc,
                float* __restrict__ probs_out, float* __restrict__ context, int t)
{
    __shared__ float p[SS];
    int b = blockIdx.x;
    if (threadIdx.x < SS) p[threadIdx.x] = scores[b * SS + threadIdx.x];
    __syncthreads();
    if (threadIdx.x == 0) {
        float m = -3.4e38f;
        for (int s = 0; s < SS; ++s) m = fmaxf(m, p[s]);
        float sum = 0.0f;
        for (int s = 0; s < SS; ++s) { p[s] = expf(p[s] - m); sum += p[s]; }
        float inv = 1.0f / sum;
        for (int s = 0; s < SS; ++s) p[s] *= inv;
    }
    __syncthreads();
    if (threadIdx.x < SS)
        probs_out[((size_t)b * TT + t) * SS + threadIdx.x] = p[threadIdx.x];
    for (int e = threadIdx.x; e < ENC; e += blockDim.x) {
        const float* eb = enc + (size_t)b * SS * ENC + e;
        float acc = 0.0f;
        for (int s = 0; s < SS; ++s) acc += p[s] * eb[(size_t)s * ENC];
        context[(size_t)b * ENC + e] = acc;
    }
}

// av_in bf16 [B][H+ENC] = [h_new, context]
__global__ void __launch_bounds__(256)
pack_avin(const float* __restrict__ h, const float* __restrict__ context,
          __bf16* __restrict__ avin)
{
    int idx = blockIdx.x * blockDim.x + threadIdx.x;
    if (idx >= BB * (HH + ENC)) return;
    int b = idx / (HH + ENC), k = idx % (HH + ENC);
    float v = (k < HH) ? h[(size_t)b * HH + k]
                       : context[(size_t)b * ENC + (k - HH)];
    avin[idx] = (__bf16)v;
}

// av = tanh(avpre + bias); write f32 att_vectors slice, a (input feed), bf16 row
__global__ void __launch_bounds__(256)
av_epilogue(const float* __restrict__ avpre, const float* __restrict__ avb,
            float* __restrict__ a, float* __restrict__ out_attvec,
            __bf16* __restrict__ avrows, int t)
{
    int idx = blockIdx.x * blockDim.x + threadIdx.x;
    if (idx >= BB * HH) return;
    int b = idx / HH, j = idx % HH;
    float v = tanhf(avpre[idx] + avb[j]);
    a[idx] = v;
    size_t o = ((size_t)b * TT + t) * HH + j;
    out_attvec[o] = v;
    avrows[o] = (__bf16)v;
}

__global__ void __launch_bounds__(256)
copy_hidden(const float* __restrict__ h, float* __restrict__ out_hidden)
{
    int idx = blockIdx.x * blockDim.x + threadIdx.x;
    if (idx < BB * HH) out_hidden[idx] = h[idx];
}

// ---------------------------------------------------------------------------
// Host orchestration
// ---------------------------------------------------------------------------
static inline size_t align256(size_t x) { return (x + 255) & ~(size_t)255; }

extern "C" void kernel_launch(void* const* d_in, const int* in_sizes, int n_in,
                              void* d_out, int out_size, void* d_ws, size_t ws_size,
                              hipStream_t stream)
{
    (void)in_sizes; (void)n_in; (void)out_size; (void)ws_size;

    // Inputs (setup_inputs order)
    const float* trg_embed   = (const float*)d_in[0];   // [B,T,EMB]
    const float* enc_out     = (const float*)d_in[1];   // [B,S,ENC]
    /* encoder_hidden d_in[2] unused (bridge=False -> h0 = zeros) */
    const unsigned char* src_mask = (const unsigned char*)d_in[3]; // [B,S] bool
    /* unrol_steps d_in[4]: hardcoded T=32 */
    const float* W_ih   = (const float*)d_in[5];   // [3H, EMB+H]
    const float* b_ih   = (const float*)d_in[6];
    const float* W_hh   = (const float*)d_in[7];   // [3H, H]
    const float* b_hh   = (const float*)d_in[8];
    const float* att_key_W   = (const float*)d_in[9];   // [ENC, H]
    const float* att_query_W = (const float*)d_in[10];  // [H, H]
    const float* att_energy_v= (const float*)d_in[11];  // [H]
    const float* att_vec_W   = (const float*)d_in[12];  // [H, H+ENC]
    const float* att_vec_b   = (const float*)d_in[13];  // [H]
    const float* out_W       = (const float*)d_in[14];  // [V, H]

    // Outputs, flat-concatenated
    float* out_outputs = (float*)d_out;                              // B*T*V
    float* out_hidden  = out_outputs + (size_t)BB * TT * VV;         // B*H
    float* out_probs   = out_hidden + (size_t)BB * HH;               // B*T*S
    float* out_attvec  = out_probs + (size_t)BB * TT * SS;           // B*T*H

    // Workspace carve-up (~123 MB total)
    char* w = (char*)d_ws;
    auto alloc = [&](size_t bytes) { void* p = (void*)w; w += align256(bytes); return p; };
    __bf16* outW_bf16 = (__bf16*)alloc((size_t)VV * HH * 2);          // [V][H]   T-form
    __bf16* Wih_bf16  = (__bf16*)alloc((size_t)3 * HH * (EMB + HH) * 2);
    __bf16* Whh_bf16  = (__bf16*)alloc((size_t)3 * HH * HH * 2);
    __bf16* Wav_bf16  = (__bf16*)alloc((size_t)HH * (HH + ENC) * 2);
    __bf16* WqT_bf16  = (__bf16*)alloc((size_t)HH * HH * 2);          // [H][H]   transposed
    __bf16* WkT_bf16  = (__bf16*)alloc((size_t)HH * ENC * 2);         // [H][ENC] transposed
    __bf16* enc_bf16  = (__bf16*)alloc((size_t)BB * SS * ENC * 2);
    float*  projk     = (float*)alloc((size_t)BB * SS * HH * 4);
    float*  gi        = (float*)alloc((size_t)BB * 3 * HH * 4);
    float*  gh        = (float*)alloc((size_t)BB * 3 * HH * 4);
    float*  h_f32     = (float*)alloc((size_t)BB * HH * 4);
    float*  a_f32     = (float*)alloc((size_t)BB * HH * 4);
    __bf16* x_bf16    = (__bf16*)alloc((size_t)BB * (EMB + HH) * 2);
    __bf16* h_bf16    = (__bf16*)alloc((size_t)BB * HH * 2);
    __bf16* hn_bf16   = (__bf16*)alloc((size_t)BB * HH * 2);
    float*  pq        = (float*)alloc((size_t)BB * HH * 4);
    float*  scores    = (float*)alloc((size_t)BB * SS * 4);
    float*  context   = (float*)alloc((size_t)BB * ENC * 4);
    __bf16* avin_bf16 = (__bf16*)alloc((size_t)BB * (HH + ENC) * 2);
    float*  avpre     = (float*)alloc((size_t)BB * HH * 4);
    __bf16* avrows    = (__bf16*)alloc((size_t)BB * TT * HH * 2);     // [B*T][H]

    // ---- one-time prep: bf16 weight copies / transposes ----
    {
        size_t n;
        n = (size_t)VV * HH;
        convert_f32_bf16<<<2048, 256, 0, stream>>>(out_W, outW_bf16, n);
        n = (size_t)3 * HH * (EMB + HH);
        convert_f32_bf16<<<1024, 256, 0, stream>>>(W_ih, Wih_bf16, n);
        n = (size_t)3 * HH * HH;
        convert_f32_bf16<<<1024, 256, 0, stream>>>(W_hh, Whh_bf16, n);
        n = (size_t)HH * (HH + ENC);
        convert_f32_bf16<<<1024, 256, 0, stream>>>(att_vec_W, Wav_bf16, n);
        n = (size_t)BB * SS * ENC;
        convert_f32_bf16<<<1024, 256, 0, stream>>>(enc_out, enc_bf16, n);
        transpose_f32_bf16<<<dim3(HH / 32, HH / 32), 256, 0, stream>>>(
            att_query_W, WqT_bf16, HH, HH);
        transpose_f32_bf16<<<dim3(HH / 32, ENC / 32), 256, 0, stream>>>(
            att_key_W, WkT_bf16, ENC, HH);
    }
    hipMemsetAsync(h_f32, 0, (size_t)BB * HH * 4, stream);
    hipMemsetAsync(a_f32, 0, (size_t)BB * HH * 4, stream);

    // proj_keys[b,s,:] = enc_out[b,s,:] @ att_key_W  -> [4096, 1024], K=ENC
    gemm_bf16_tn<<<dim3(HH / 256, (BB * SS) / 64), 128, 0, stream>>>(
        enc_bf16, WkT_bf16, projk, BB * SS, HH, ENC);

    // ---- sequential decode, T steps ----
    for (int t = 0; t < TT; ++t) {
        pack_x<<<(BB * (EMB + 2 * HH) + 255) / 256, 256, 0, stream>>>(
            trg_embed, a_f32, h_f32, x_bf16, h_bf16, t);
        // gi = x @ W_ih^T : M=64, N=3072, K=1536
        gemm_bf16_tn<<<dim3((3 * HH) / 256, 1), 128, 0, stream>>>(
            x_bf16, Wih_bf16, gi, BB, 3 * HH, EMB + HH);
        // gh = h @ W_hh^T : M=64, N=3072, K=1024
        gemm_bf16_tn<<<dim3((3 * HH) / 256, 1), 128, 0, stream>>>(
            h_bf16, Whh_bf16, gh, BB, 3 * HH, HH);
        gru_update<<<(BB * HH + 255) / 256, 256, 0, stream>>>(
            gi, gh, b_ih, b_hh, h_f32, hn_bf16);
        // pq = h_new @ att_query_W : M=64, N=1024, K=1024
        gemm_bf16_tn<<<dim3(HH / 256, 1), 128, 0, stream>>>(
            hn_bf16, WqT_bf16, pq, BB, HH, HH);
        energy_scores<<<(BB * SS * 32) / 256, 256, 0, stream>>>(
            projk, pq, att_energy_v, src_mask, scores);
        softmax_context<<<BB, 256, 0, stream>>>(
            scores, enc_out, out_probs, context, t);
        pack_avin<<<(BB * (HH + ENC) + 255) / 256, 256, 0, stream>>>(
            h_f32, context, avin_bf16);
        // avpre = [h_new, ctx] @ att_vec_W^T : M=64, N=1024, K=2048
        gemm_bf16_tn<<<dim3(HH / 256, 1), 128, 0, stream>>>(
            avin_bf16, Wav_bf16, avpre, BB, HH, HH + ENC);
        av_epilogue<<<(BB * HH + 255) / 256, 256, 0, stream>>>(
            avpre, att_vec_b, a_f32, out_attvec, avrows, t);
    }

    copy_hidden<<<(BB * HH + 255) / 256, 256, 0, stream>>>(h_f32, out_hidden);

    // outputs = att_vectors @ out_W^T : M=2048, N=32000, K=1024 (dominant GEMM)
    gemm_bf16_tn<<<dim3(VV / 256, (BB * TT) / 64), 128, 0, stream>>>(
        avrows, outW_bf16, out_outputs, BB * TT, VV, HH);
}
